// HeteroGNN_5918464934161
// MI455X (gfx1250) — matrix operand order repaired and checked by reading, not compiled, not run on hardware
//
#include <hip/hip_runtime.h>
#include <hip/hip_bf16.h>

typedef __bf16 bf16_t;
typedef __attribute__((ext_vector_type(16))) __bf16 v16bf;
typedef __attribute__((ext_vector_type(8)))  __bf16 v8bf;
typedef __attribute__((ext_vector_type(8)))  float  v8f;

#define LDS_STRIDE 136   // 128 + 8 pad -> 272B row stride: lane n at n*16B mod 256B, bank-conflict-free

__device__ __forceinline__ float gelu_exact(float v) {
    // torch F.gelu default: exact erf
    return 0.5f * v * (1.0f + erff(v * 0.70710678118654752440f));
}

// ---------------------------------------------------------------------------
// Convert + transpose weights: W[mat][in][out] (fp32) -> Wt[mat][out][in] (bf16)
// so B-fragments (K-major slices per output column) are contiguous.
// ---------------------------------------------------------------------------
__global__ __launch_bounds__(256) void hgnn_cvt_w(const float* __restrict__ W,
                                                  bf16_t* __restrict__ Wt,
                                                  int total) {
    int i = blockIdx.x * 256 + threadIdx.x;
    if (i >= total) return;
    int mat = i >> 14;          // 128*128 per matrix
    int o   = (i >> 7) & 127;   // output channel (row of Wt)
    int in  = i & 127;          // input channel  (col of Wt)
    Wt[i] = (bf16_t)W[(mat << 14) + in * 128 + o];
}

__global__ __launch_bounds__(256) void hgnn_zero(float* __restrict__ p, int n) {
    int i = blockIdx.x * 256 + threadIdx.x;
    if (i < n) p[i] = 0.0f;
}

__global__ __launch_bounds__(256) void hgnn_residual_gelu(float* __restrict__ x,
                                                          const float* __restrict__ o,
                                                          int n) {
    int i = blockIdx.x * 256 + threadIdx.x;
    if (i < n) x[i] += gelu_exact(o[i]);
}

// Stage the 128x128 bf16 weight matrix into LDS with padded rows.
// 256 threads x 16B, 8 iterations = 32KB.
__device__ __forceinline__ void stage_weights(bf16_t* __restrict__ wlds,
                                              const bf16_t* __restrict__ Wt) {
    const int tid = threadIdx.x;
    const int c = tid & 15;          // 16B chunk within a 256B row
    #pragma unroll
    for (int r = tid >> 4; r < 128; r += 16) {
        ((uint4*)(wlds + r * LDS_STRIDE))[c] = ((const uint4*)(Wt + r * 128))[c];
    }
    __syncthreads();
}

// Read one 32x16 B-fragment from LDS: lane holds column N = lane&15,
// K = k0 + (lane>>4)*16 .. +15  (two conflict-free ds_load_b128).
__device__ __forceinline__ v16bf load_bfrag(const bf16_t* __restrict__ wlds,
                                            int nt, int nn, int half, int k0) {
    const bf16_t* p = wlds + (nt * 16 + nn) * LDS_STRIDE + k0 + half * 16;
    v8bf lo = *(const v8bf*)p;
    v8bf hi = *(const v8bf*)(p + 8);
    return __builtin_shufflevector(lo, hi, 0,1,2,3,4,5,6,7,8,9,10,11,12,13,14,15);
}

// ---------------------------------------------------------------------------
// Generic [rows,128] @ [128,128] + bias GEMM in bf16 WMMA, fp32 accumulate.
//   C = (gelu?)( alpha*A @ Wt^T + bias + addC )   [+= if accumulate]
// SCALE: alpha = 1 + *epsP folded into the A->bf16 conversion.
// Block = 256 threads = 8 waves; each wave computes a 16x128 output strip.
// Weight matrix staged once per block in LDS.
// ---------------------------------------------------------------------------
template <bool SCALE>
__global__ __launch_bounds__(256) void hgnn_gemm128_wmma(
    const float*  __restrict__ A,     // [rows,128] fp32
    const bf16_t* __restrict__ Wt,    // [128(out),128(in)] bf16 (transposed)
    const float*  __restrict__ bias,  // [128]
    const float*  __restrict__ addC,  // [rows,128] or null
    const float*  __restrict__ epsP,  // scalar, used when SCALE
    float*        __restrict__ C,     // [rows,128]
    int rows, int flags)              // flags: bit0 = gelu, bit1 = accumulate
{
    __shared__ bf16_t wlds[128 * LDS_STRIDE];
    stage_weights(wlds, Wt);

    const int lane = threadIdx.x & 31;
    const int wave = threadIdx.x >> 5;
    const int row0 = blockIdx.x * 128 + wave * 16;
    if (row0 >= rows) return;                 // uniform per wave: EXEC stays full
    const int nn   = lane & 15;
    const int half = lane >> 4;
    float alpha = 1.0f;
    if (SCALE) alpha = 1.0f + epsP[0];

    v8f acc[8] = {};

    // A-fragment source row for this lane (ISA 16-bit A 16x32 layout: lane holds M = lane&15)
    const int ra = min(row0 + nn, rows - 1);
    const float* Arow = A + (size_t)ra * 128;

    #pragma unroll
    for (int ks = 0; ks < 4; ++ks) {
        const int k0 = ks * 32;
        v16bf a;
        #pragma unroll
        for (int i = 0; i < 8; ++i) {
            float lo = Arow[k0 + half * 16 + i];
            float hi = Arow[k0 + half * 16 + 8 + i];
            if (SCALE) { lo *= alpha; hi *= alpha; }
            a[i]     = (bf16_t)lo;
            a[8 + i] = (bf16_t)hi;
        }
        #pragma unroll
        for (int nt = 0; nt < 8; ++nt) {
            const v16bf b = load_bfrag(wlds, nt, nn, half, k0);
            acc[nt] = __builtin_amdgcn_wmma_f32_16x16x32_bf16(
                false, a, false, b, (short)0, acc[nt], false, false);
        }
    }

    // C/D layout: lane holds column N = lane&15, rows M = 8*(lane>>4) + j
    const int mb = half * 8;
    #pragma unroll
    for (int nt = 0; nt < 8; ++nt) {
        const int c = nt * 16 + nn;
        const float bv = bias[c];
        #pragma unroll
        for (int j = 0; j < 8; ++j) {
            const int r = row0 + mb + j;
            if (r < rows) {
                float v = acc[nt][j] + bv;
                if (addC)      v += addC[(size_t)r * 128 + c];
                if (flags & 1) v = gelu_exact(v);
                if (flags & 2) C[(size_t)r * 128 + c] += v;
                else           C[(size_t)r * 128 + c]  = v;
            }
        }
    }
}

// ---------------------------------------------------------------------------
// Fused GINE edge kernel: for a 16-edge tile,
//   e   = ea @ We (WMMA, LDS-staged weights)   (message tensor never hits HBM)
//   m   = gelu(xs[src] + e + be) * ew
//   agg[dst] += m                              (global_atomic_add_f32 scatter)
// ---------------------------------------------------------------------------
__global__ __launch_bounds__(256) void hgnn_edge128_wmma(
    const float*  __restrict__ ea,    // [E,128] fp32
    const bf16_t* __restrict__ Wet,   // [128,128] bf16 transposed
    const float*  __restrict__ be,    // [128]
    const int*    __restrict__ src,   // [E]
    const int*    __restrict__ dst,   // [E]
    const float*  __restrict__ ew,    // [E]
    const float*  __restrict__ xs,    // [Nsrc,128] = lin_src output
    float*        __restrict__ agg,   // [Ndst,128]
    int E)
{
    __shared__ bf16_t wlds[128 * LDS_STRIDE];
    stage_weights(wlds, Wet);

    const int lane = threadIdx.x & 31;
    const int wave = threadIdx.x >> 5;
    const int e0   = blockIdx.x * 128 + wave * 16;
    if (e0 >= E) return;
    const int nn   = lane & 15;
    const int half = lane >> 4;

    v8f acc[8] = {};

    const int eL = min(e0 + nn, E - 1);
    const float* Arow = ea + (size_t)eL * 128;

    #pragma unroll
    for (int ks = 0; ks < 4; ++ks) {
        const int k0 = ks * 32;
        v16bf a;
        #pragma unroll
        for (int i = 0; i < 8; ++i) {
            a[i]     = (bf16_t)Arow[k0 + half * 16 + i];
            a[8 + i] = (bf16_t)Arow[k0 + half * 16 + 8 + i];
        }
        #pragma unroll
        for (int nt = 0; nt < 8; ++nt) {
            const v16bf b = load_bfrag(wlds, nt, nn, half, k0);
            acc[nt] = __builtin_amdgcn_wmma_f32_16x16x32_bf16(
                false, a, false, b, (short)0, acc[nt], false, false);
        }
    }

    const int mb = half * 8;
    #pragma unroll
    for (int j = 0; j < 8; ++j) {
        const int e = e0 + mb + j;
        if (e < E) {
            const int   s = src[e];
            const int   d = dst[e];
            const float w = ew[e];
            #pragma unroll
            for (int nt = 0; nt < 8; ++nt) {
                const int c = nt * 16 + nn;
                float v = gelu_exact(acc[nt][j] + be[c] + xs[(size_t)s * 128 + c]) * w;
                atomicAdd(&agg[(size_t)d * 128 + c], v);
            }
        }
    }
}

// ---------------------------------------------------------------------------
// Host orchestration
// ---------------------------------------------------------------------------
extern "C" void kernel_launch(void* const* d_in, const int* in_sizes, int n_in,
                              void* d_out, int out_size, void* d_ws, size_t ws_size,
                              hipStream_t stream) {
    (void)n_in; (void)out_size; (void)ws_size;
    const int HID = 128;

    const float* x_base = (const float*)d_in[0];
    const float* x_cent = (const float*)d_in[1];
    const int NB = in_sizes[0] / HID;
    const int NC = in_sizes[1] / HID;

    // edge type order: 0=bb 1=bc 2=cc 3=cb ; dict order bb,bc,cc,cb at slots 2..13
    const int*   ei[4] = {(const int*)d_in[2], (const int*)d_in[5], (const int*)d_in[8], (const int*)d_in[11]};
    const float* ea[4] = {(const float*)d_in[3], (const float*)d_in[6], (const float*)d_in[9], (const float*)d_in[12]};
    const float* ew[4] = {(const float*)d_in[4], (const float*)d_in[7], (const float*)d_in[10], (const float*)d_in[13]};
    int Ecnt[4] = {in_sizes[2] / 2, in_sizes[5] / 2, in_sizes[8] / 2, in_sizes[11] / 2};

    const float* Wsrc = (const float*)d_in[14];
    const float* bsrc = (const float*)d_in[15];
    const float* Wdst = (const float*)d_in[16];
    const float* bdst = (const float*)d_in[17];
    const float* eps  = (const float*)d_in[18];
    const float* We   = (const float*)d_in[19];
    const float* be   = (const float*)d_in[20];
    const float* Wm1  = (const float*)d_in[21];
    const float* bm1  = (const float*)d_in[22];
    const float* Wm2  = (const float*)d_in[23];
    const float* bm2  = (const float*)d_in[24];

    // workspace carve-up (256B aligned)
    char* wp = (char*)d_ws;
    auto carve = [&](size_t bytes) -> void* {
        void* q = (void*)wp;
        wp += (bytes + 255) & ~(size_t)255;
        return q;
    };
    float* xs  = (float*)carve((size_t)NB * HID * 4);   // lin_src out; reused as MLP hidden t1
    float* agg = (float*)carve((size_t)NB * HID * 4);   // scatter target; reused in-place as h
    float* ob  = (float*)carve((size_t)NB * HID * 4);   // base conv-sum
    float* oc  = (float*)carve((size_t)NC * HID * 4);   // cent conv-sum
    bf16_t* tw[5];                                      // bf16-transposed weights: Wsrc,Wdst,We,Wm1,Wm2
    for (int i = 0; i < 5; ++i) tw[i] = (bf16_t*)carve((size_t)2 * 4 * HID * HID * 2);

    float* xb = (float*)d_out;                 // evolving base features
    float* xc = xb + (size_t)NB * HID;         // evolving centroid features

    hipMemcpyAsync(xb, x_base, (size_t)NB * HID * 4, hipMemcpyDeviceToDevice, stream);
    hipMemcpyAsync(xc, x_cent, (size_t)NC * HID * 4, hipMemcpyDeviceToDevice, stream);

    // convert all 5 weight tensors (2 layers x 4 types each)
    const float* wsF[5] = {Wsrc, Wdst, We, Wm1, Wm2};
    const int wtotal = 2 * 4 * HID * HID;
    for (int i = 0; i < 5; ++i)
        hgnn_cvt_w<<<(wtotal + 255) / 256, 256, 0, stream>>>(wsF[i], tw[i], wtotal);

    auto gblocks = [](int rows) { return (unsigned)((rows + 127) / 128); };
    auto launch_gemm = [&](const float* A, const bf16_t* Wt, const float* bias,
                           const float* addC, const float* epsP, float* C,
                           int rows, int flags) {
        if (epsP)
            hgnn_gemm128_wmma<true><<<gblocks(rows), 256, 0, stream>>>(
                A, Wt, bias, addC, epsP, C, rows, flags);
        else
            hgnn_gemm128_wmma<false><<<gblocks(rows), 256, 0, stream>>>(
                A, Wt, bias, addC, nullptr, C, rows, flags);
    };

    for (int l = 0; l < 2; ++l) {
        // conv schedule: (bb -> ob write), (cb -> ob accum), (bc -> oc write), (cc -> oc accum)
        const int ts[4] = {0, 3, 1, 2};
        for (int k = 0; k < 4; ++k) {
            const int t = ts[k];
            const int srcN = (t == 0 || t == 1) ? NB : NC;   // bb/bc source = base
            const int dstN = (t == 0 || t == 3) ? NB : NC;   // bb/cb dest   = base
            const float* srcX = (t == 0 || t == 1) ? xb : xc;
            const float* dstX = (t == 0 || t == 3) ? xb : xc;
            float* outB = (t == 0 || t == 3) ? ob : oc;
            const int accum = (k & 1) ? 2 : 0;
            const int wi = l * 4 + t;
            const size_t wo = (size_t)wi * HID * HID;
            const int E = Ecnt[t];

            // xs = srcX @ Wsrc + bsrc
            launch_gemm(srcX, tw[0] + wo, bsrc + wi * HID, nullptr, nullptr, xs, srcN, 0);

            // agg = 0
            hgnn_zero<<<(dstN * HID + 255) / 256, 256, 0, stream>>>(agg, dstN * HID);

            // fused: e = ea@We+be ; m = gelu(xs[src]+e)*ew ; agg[dst] += m
            hgnn_edge128_wmma<<<(unsigned)((E + 127) / 128), 256, 0, stream>>>(
                ea[t], tw[2] + wo, be + wi * HID, ei[t], ei[t] + E, ew[t], xs, agg, E);

            // h = agg + ((1+eps)*dstX) @ Wdst + bdst     (in-place over agg)
            launch_gemm(dstX, tw[1] + wo, bdst + wi * HID, agg, eps + wi, agg, dstN, 0);

            // t1 = gelu(h @ Wm1 + bm1)                   (t1 aliases xs)
            launch_gemm(agg, tw[3] + wo, bm1 + wi * HID, nullptr, nullptr, xs, dstN, 1);

            // outB (=|+=) t1 @ Wm2 + bm2
            launch_gemm(xs, tw[4] + wo, bm2 + wi * HID, nullptr, nullptr, outB, dstN, accum);
        }
        // residuals: x += gelu(o)
        hgnn_residual_gelu<<<(NB * HID + 255) / 256, 256, 0, stream>>>(xb, ob, NB * HID);
        hgnn_residual_gelu<<<(NC * HID + 255) / 256, 256, 0, stream>>>(xc, oc, NC * HID);
    }
}
